// HyperGraph1_50371376447884
// MI455X (gfx1250) — matrix operand order repaired and checked by reading, not compiled, not run on hardware
//
#include <hip/hip_runtime.h>

#define NFEAT   256
#define NCLASS  64
#define N_EDGES 20000
#define KBLOCKS (NFEAT / 32)   // 8
#define NTILES  (NCLASS / 16)  // 4
// fragment elements per (kblock, ntile): 32 lanes * 16 bf16
#define WFRAG_ELEMS (KBLOCKS * NTILES * 32 * 16)   // 16384

typedef __attribute__((ext_vector_type(16))) __bf16 v16bf;
typedef __attribute__((ext_vector_type(8)))  float  v8f;

// ---------------- bf16 split helpers (hardware RNE converts) ----------------
static __device__ __forceinline__ __bf16 cvt_bf16(float f) { return (__bf16)f; }
static __device__ __forceinline__ float  cvt_f32(__bf16 b) { return (float)b; }

// ---------------------------- utility kernels -------------------------------
__global__ void zero_f32_kernel(float* __restrict__ p, long long n) {
  long long i = (long long)blockIdx.x * blockDim.x + threadIdx.x;
  long long stride = (long long)gridDim.x * blockDim.x;
  for (; i < n; i += stride) p[i] = 0.0f;
}

__global__ void degree_kernel(const int* __restrict__ node_idx,
                              const int* __restrict__ edge_idx,
                              float* __restrict__ Ddeg, float* __restrict__ Bdeg,
                              int n_inc) {
  int i = blockIdx.x * blockDim.x + threadIdx.x;
  if (i < n_inc) {
    unsafeAtomicAdd(&Ddeg[node_idx[i]], 1.0f);
    unsafeAtomicAdd(&Bdeg[edge_idx[i]], 1.0f);
  }
}

// ---- pre-convert + pre-swizzle W into per-lane B fragments (bf16 hi/lo) ----
// Fragment layout: idx = (((kb/32)*NTILES + nt)*32 + lane)*16 + e
// holds W[k][n] with k = kb + (e/8)*16 + (lane/16)*8 + e%8, n = nt*16 + lane%16.
__global__ void prep_wfrag_kernel(const float* __restrict__ W,
                                  __bf16* __restrict__ Wfrag_hi,
                                  __bf16* __restrict__ Wfrag_lo) {
  int t = blockIdx.x * blockDim.x + threadIdx.x;
  if (t >= WFRAG_ELEMS) return;
  const int e    = t & 15;
  const int lane = (t >> 4) & 31;
  const int nt   = (t >> 9) & (NTILES - 1);
  const int kbi  = t >> 11;
  const int k = kbi * 32 + (e & 8) * 2 + (lane >> 4) * 8 + (e & 7);
  const int n = nt * 16 + (lane & 15);
  const float f = W[k * NCLASS + n];
  const __bf16 hi = cvt_bf16(f);
  Wfrag_hi[t] = hi;
  Wfrag_lo[t] = cvt_bf16(f - cvt_f32(hi));
}

// ------------------- xw = x @ W, bf16x3 WMMA, fp32 accumulate ---------------
// 8 waves per block; each wave owns one 16-row M tile and all 4 N tiles.
__global__ void __launch_bounds__(256)
gemm_xw_wmma(const float* __restrict__ x,
             const __bf16* __restrict__ Wfrag_hi,
             const __bf16* __restrict__ Wfrag_lo,
             float* __restrict__ xw, int n_nodes, int total_mtiles) {
  const int lane  = threadIdx.x & 31;
  const int wave  = threadIdx.x >> 5;
  const int mtile = blockIdx.x * 8 + wave;
  if (mtile >= total_mtiles) return;      // wave-uniform: EXEC stays all-1s

  const int half = lane >> 4;             // 0: lanes 0-15, 1: lanes 16-31
  const int l16  = lane & 15;
  const int m0   = mtile * 16;

  int row = m0 + l16;
  if (row >= n_nodes) row = n_nodes - 1;  // clamp (no predication around WMMA)

  v8f acc[NTILES];
#pragma unroll
  for (int nt = 0; nt < NTILES; ++nt) acc[nt] = v8f{};

  for (int kbi = 0; kbi < KBLOCKS; ++kbi) {
    const int kb = kbi * 32;
    // ---- A fragment: two contiguous 8-float chunks -> b128 loads ----
    // element e<8 : K = kb + half*8 + e ; e>=8 : K = kb + 16 + half*8 + (e-8)
    const float* xrow = x + row * NFEAT + kb + half * 8;
    float af[16];
#pragma unroll
    for (int j = 0; j < 8; ++j) af[j] = xrow[j];
#pragma unroll
    for (int j = 0; j < 8; ++j) af[8 + j] = xrow[16 + j];

    v16bf a_hi, a_lo;
#pragma unroll
    for (int e = 0; e < 16; ++e) {
      const __bf16 hi = cvt_bf16(af[e]);
      a_hi[e] = hi;
      a_lo[e] = cvt_bf16(af[e] - cvt_f32(hi));
    }

#pragma unroll
    for (int nt = 0; nt < NTILES; ++nt) {
      // ---- B fragments: one contiguous 32B vector load each ----
      const int fi = ((kbi * NTILES + nt) * 32 + lane) * 16;
      const v16bf b_hi = *(const v16bf*)(Wfrag_hi + fi);
      const v16bf b_lo = *(const v16bf*)(Wfrag_lo + fi);
      v8f c = acc[nt];
      c = __builtin_amdgcn_wmma_f32_16x16x32_bf16(false, a_hi, false, b_hi,
                                                  (short)0, c, false, false);
      c = __builtin_amdgcn_wmma_f32_16x16x32_bf16(false, a_hi, false, b_lo,
                                                  (short)0, c, false, false);
      c = __builtin_amdgcn_wmma_f32_16x16x32_bf16(false, a_lo, false, b_hi,
                                                  (short)0, c, false, false);
      acc[nt] = c;
    }
  }

  // ---- store C/D: N = l16, M = v + 8*half ----
  float* outp = xw + (m0 + half * 8) * NCLASS + l16;
  if (m0 + 16 <= n_nodes) {
    // fast path (taken by every full tile): unguarded store stream
#pragma unroll
    for (int nt = 0; nt < NTILES; ++nt) {
#pragma unroll
      for (int v = 0; v < 8; ++v) {
        outp[v * NCLASS + nt * 16] = acc[nt][v];
      }
    }
  } else {
    // ragged last tile only
#pragma unroll
    for (int nt = 0; nt < NTILES; ++nt) {
#pragma unroll
      for (int v = 0; v < 8; ++v) {
        const int m = m0 + v + half * 8;
        if (m < n_nodes) xw[m * NCLASS + nt * 16 + l16] = acc[nt][v];
      }
    }
  }
}

// ----------------- scatter phases (wave-uniform index loads) ----------------
__global__ void scatter_node_to_edge(const int* __restrict__ node_idx,
                                     const int* __restrict__ edge_idx,
                                     const float* __restrict__ xw,
                                     float* __restrict__ e_feat, long long total) {
  long long t = (long long)blockIdx.x * blockDim.x + threadIdx.x;
  if (t >= total) return;
  const int i = (int)(t >> 6);
  const int c = (int)(t & 63);
  const int n = node_idx[i];
  const int e = edge_idx[i];
  unsafeAtomicAdd(&e_feat[e * NCLASS + c], xw[n * NCLASS + c]);
}

__global__ void scale_efeat(float* __restrict__ e_feat,
                            const float* __restrict__ Bdeg, int total) {
  int t = blockIdx.x * blockDim.x + threadIdx.x;
  if (t < total) {
    const float bb = Bdeg[t >> 6];
    const float binv = bb > 0.0f ? 1.0f / bb : 0.0f;
    e_feat[t] *= binv;
  }
}

__global__ void scatter_edge_to_node(const int* __restrict__ node_idx,
                                     const int* __restrict__ edge_idx,
                                     const float* __restrict__ e_feat,
                                     float* __restrict__ out, long long total) {
  long long t = (long long)blockIdx.x * blockDim.x + threadIdx.x;
  if (t >= total) return;
  const int i = (int)(t >> 6);
  const int c = (int)(t & 63);
  const int n = node_idx[i];
  const int e = edge_idx[i];
  unsafeAtomicAdd(&out[n * NCLASS + c], e_feat[e * NCLASS + c]);
}

__global__ void finalize_kernel(float* __restrict__ out,
                                const float* __restrict__ Ddeg,
                                const float* __restrict__ bias, long long total) {
  long long t = (long long)blockIdx.x * blockDim.x + threadIdx.x;
  if (t >= total) return;
  const int n = (int)(t >> 6);
  const int c = (int)(t & 63);
  const float d = Ddeg[n];
  const float dinv = d > 0.0f ? 1.0f / d : 0.0f;
  const float v = out[t] * dinv + bias[c];
  out[t] = v > 0.0f ? v : 0.0f;
}

// ------------------------------- launcher -----------------------------------
extern "C" void kernel_launch(void* const* d_in, const int* in_sizes, int n_in,
                              void* d_out, int out_size, void* d_ws, size_t ws_size,
                              hipStream_t stream) {
  const float* x    = (const float*)d_in[0];
  const int*   adj  = (const int*)d_in[1];
  const float* W    = (const float*)d_in[2];
  const float* bias = (const float*)d_in[3];
  float* out = (float*)d_out;

  const long long n_inc   = in_sizes[1] / 2;
  const long long n_nodes = in_sizes[0] / NFEAT;
  const int* node_idx = adj;
  const int* edge_idx = adj + n_inc;

  // workspace layout (floats): xw | e_feat | D | B | Wfrag_hi | Wfrag_lo
  float* ws     = (float*)d_ws;
  float* xw     = ws;
  float* e_feat = xw + n_nodes * NCLASS;
  float* Ddeg   = e_feat + (long long)N_EDGES * NCLASS;
  float* Bdeg   = Ddeg + n_nodes;
  __bf16* Wfrag_hi = (__bf16*)(Bdeg + N_EDGES);
  __bf16* Wfrag_lo = Wfrag_hi + WFRAG_ELEMS;

  // 1) zero accumulators every call (atomics accumulate)
  const long long zero_ws_n = (long long)N_EDGES * NCLASS + n_nodes + N_EDGES;
  zero_f32_kernel<<<2048, 256, 0, stream>>>(e_feat, zero_ws_n);
  zero_f32_kernel<<<4096, 256, 0, stream>>>(out, n_nodes * NCLASS);

  // 2) degrees + W fragment prep (independent, same stream)
  degree_kernel<<<(int)((n_inc + 255) / 256), 256, 0, stream>>>(
      node_idx, edge_idx, Ddeg, Bdeg, (int)n_inc);
  prep_wfrag_kernel<<<(WFRAG_ELEMS + 255) / 256, 256, 0, stream>>>(
      W, Wfrag_hi, Wfrag_lo);

  // 3) xw = x @ W via WMMA (bf16x3)
  const int m_tiles = (int)((n_nodes + 15) / 16);
  gemm_xw_wmma<<<(m_tiles + 7) / 8, 256, 0, stream>>>(
      x, Wfrag_hi, Wfrag_lo, xw, (int)n_nodes, m_tiles);

  // 4) node -> hyperedge scatter, 5) scale by B^-1
  const long long total_inc = n_inc * NCLASS;
  const int inc_blocks = (int)((total_inc + 255) / 256);
  scatter_node_to_edge<<<inc_blocks, 256, 0, stream>>>(node_idx, edge_idx, xw,
                                                       e_feat, total_inc);
  scale_efeat<<<(N_EDGES * NCLASS + 255) / 256, 256, 0, stream>>>(
      e_feat, Bdeg, N_EDGES * NCLASS);

  // 6) hyperedge -> node scatter into d_out, 7) finalize (D^-1, +b, ReLU)
  scatter_edge_to_node<<<inc_blocks, 256, 0, stream>>>(node_idx, edge_idx,
                                                       e_feat, out, total_inc);
  const long long total_out = n_nodes * NCLASS;
  finalize_kernel<<<(int)((total_out + 255) / 256), 256, 0, stream>>>(
      out, Ddeg, bias, total_out);
}
